// ForcedAlignmentTaskModel_79800492359910
// MI455X (gfx1250) — compile-verified, matrix operand order checked by reading, not run on hardware
//
#include <hip/hip_runtime.h>
#include <hip/hip_bf16.h>
#include <math.h>

// Problem dims (fixed by the reference)
#define Bn 32
#define Cn 256
#define Tn 4096
#define Ln 1024

// Trellis-kernel tiling
#define TILE_T 64
#define LROW   65                       // 64 data dwords + 1 pad dword (TDM pad_enable)
#define ETILE_FLOATS (Cn * LROW)        // 16640 floats per LDS tile buffer

typedef __attribute__((ext_vector_type(16))) _Float16     v16h;
typedef __attribute__((ext_vector_type(8)))  float        v8f;
typedef __attribute__((ext_vector_type(4)))  unsigned int v4u;
typedef __attribute__((ext_vector_type(8)))  int          v8i;
typedef __attribute__((ext_vector_type(4)))  int          v4i;

// ---------------------------------------------------------------------------
// Kernel 1: per-(b,t) log-sum-exp stats.  stat[b,t] = max_c x + log(sum_c exp(x-max)).
// One wave32 handles 16 consecutive t positions; the 256-class exp-sum is done
// as a WMMA row-sum against an all-ones B matrix (v_wmma_f32_16x16x32_f16).
// A-layout: lane l<16 owns row l (K slots for classes c with (c%32)<16),
// lane l>=16 owns row l-16 ((c%32)>=16).  Since B is all-ones, any K
// permutation within a row yields the same row sum.
// ---------------------------------------------------------------------------
__global__ __launch_bounds__(256) void fa_logsumexp_stats(
    const float* __restrict__ logits, float* __restrict__ stat) {
  const int wave = threadIdx.x >> 5;
  const int lane = threadIdx.x & 31;
  const int half = lane >> 4;
  const int m    = lane & 15;
  const int rowbase = (blockIdx.x * 8 + wave) * 16;   // 16 (b,t) rows per wave
  const int b  = rowbase >> 12;                       // / Tn (Tn = 4096)
  const int t0 = rowbase & (Tn - 1);
  const float* base = logits + (size_t)b * Cn * Tn;
  const int trow = t0 + m;

  // Pass 1: row max.  Each lane covers 128 classes; pair lanes l / l^16 cover all 256.
  float mx = -INFINITY;
  for (int chunk = 0; chunk < 8; ++chunk) {
#pragma unroll
    for (int i = 0; i < 16; ++i) {
      const int c = chunk * 32 + half * 16 + i;
      mx = fmaxf(mx, base[(size_t)c * Tn + trow]);
    }
  }
  mx = fmaxf(mx, __shfl_xor(mx, 16, 32));

  // Pass 2: sum(exp(x - mx)) via WMMA row-sum (D = A x ones + C), 8 chained MMAs.
  v16h ones;
#pragma unroll
  for (int i = 0; i < 16; ++i) ones[i] = (_Float16)1.0f;
  v8f acc = {0.f, 0.f, 0.f, 0.f, 0.f, 0.f, 0.f, 0.f};
  for (int chunk = 0; chunk < 8; ++chunk) {
    v16h a;
#pragma unroll
    for (int i = 0; i < 16; ++i) {
      const int c = chunk * 32 + half * 16 + i;
      a[i] = (_Float16)__expf(base[(size_t)c * Tn + trow] - mx);
    }
    acc = __builtin_amdgcn_wmma_f32_16x16x32_f16(
        /*neg_a=*/false, a, /*neg_b=*/false, ones,
        /*c_mod=*/(short)0, acc, /*reuse_a=*/false, /*reuse_b=*/false);
  }

  // C/D layout: vgpr v, lanes 0..15 -> row v; lanes 16..31 -> row v+8 (all N equal).
#pragma unroll
  for (int v = 0; v < 8; ++v) {
    const float shi = __shfl(acc[v], 16, 32);       // row v+8 sum
    if (lane == v)     stat[(size_t)b * Tn + t0 + v]     = mx + __logf(acc[v]);
    if (lane == 8 + v) stat[(size_t)b * Tn + t0 + 8 + v] = mx + __logf(shi);
  }
}

// ---------------------------------------------------------------------------
// TDM: async-load a [Cn x TILE_T] f32 tile of logits (t fastest, stride Tn)
// into LDS, padding 1 dword after every 64 dwords (LDS row stride = 65 floats
// -> conflict-free token gathers).  D# per CDNA5 ISA ch.8.
// This toolchain exposes the 6-arg builtin:
//   (uint32x4 g0, int32x8 g1, int32x4 g2, int32x4 g3, int32x8 g4, i32 cpol)
// Our tensor is 2D (tile_dim2/3 = 0), so trailing groups are zero/unused.
// ---------------------------------------------------------------------------
__device__ __forceinline__ void tdm_load_tile(unsigned int lds_byte_off,
                                              unsigned long long gaddr) {
  // Group 0: count=1 | lds_addr | global_addr[31:0] | global_addr[56:32], type=2
  v4u g0 = {1u, lds_byte_off, (unsigned int)gaddr,
            (unsigned int)(gaddr >> 32) | (2u << 30)};
  // Group 1:
  //  dw0: data_size=2 (4B) | pad_enable | pad_interval=5 (64 dw) | pad_amount=0 (1 dw)
  //  dw1: tensor_dim0[15:0]<<16            (4096)
  //  dw2: tensor_dim0[31:16] | tensor_dim1[15:0]<<16   (256)
  //  dw3: tensor_dim1[31:16] | tile_dim0<<16           (64)
  //  dw4: tile_dim1 | tile_dim2<<16                    (256, 0)
  //  dw5: tensor_dim0_stride[31:0]                     (4096)
  //  dw6,dw7: stride hi / dim1 stride (unused, 0)
  v8i g1 = {(int)0x01520000, (int)(Tn << 16), (int)(Cn << 16),
            (int)(TILE_T << 16), (int)Cn, (int)Tn, 0, 0};
  v4i gz4 = {0, 0, 0, 0};               // groups 2/3: unused for 2D tensor
  v8i gz8 = {0, 0, 0, 0, 0, 0, 0, 0};   // trailing group (clang-23 form): unused
  __builtin_amdgcn_tensor_load_to_lds(g0, g1, gz4, gz4, gz8, 0);
}

// ---------------------------------------------------------------------------
// Kernel 2: trellis DP.  One workgroup (1024 threads, 32 waves) per line b.
// prev rows double-buffered in LDS (1 barrier per step); logits staged by TDM
// with double-buffered tiles; -stat folded in after the max.
// ---------------------------------------------------------------------------
__global__ __launch_bounds__(1024) void fa_trellis(
    const float* __restrict__ logits, const int* __restrict__ tokens,
    const float* __restrict__ stat, float* __restrict__ trellis) {
  extern __shared__ char smem[];
  float* prev0 = (float*)smem;              // 1026 floats
  float* prev1 = prev0 + 1026;              // 1026 floats
  int*   toks  = (int*)(prev1 + 1026);      // 1024 ints
  float* stt   = (float*)(toks + Ln);       // TILE_T floats
  float* etile = stt + TILE_T;              // 2 * ETILE_FLOATS (byte off 12560, 16-aligned)

  const int b   = blockIdx.x;
  const int tid = threadIdx.x;
  const int jj  = tid + 1;                  // trellis column this thread owns
  const float* lbase = logits + (size_t)b * Cn * Tn;
  float* tre = trellis + (size_t)b * (Tn + 1) * (Ln + 1);

  // Row 0 init + token staging.
  toks[tid]  = tokens[(size_t)b * Ln + tid];
  prev0[jj]  = -INFINITY;
  tre[jj]    = -INFINITY;
  if (tid == 0) { prev0[0] = 0.0f; tre[0] = 0.0f; }

  float c0 = 0.0f;  // running cumsum of blank log-probs (thread 0 only)

  // Prologue: async-load tile 0 (flat LDS pointer truncates to LDS byte offset).
  if (tid == 0)
    tdm_load_tile((unsigned int)(size_t)etile, (unsigned long long)(size_t)lbase);
  __syncthreads();

  float* pA = prev0;
  float* pB = prev1;
  const int ntiles = Tn / TILE_T;
  for (int tile = 0; tile < ntiles; ++tile) {
    const int tbase = tile * TILE_T;
    if (tid < TILE_T) stt[tid] = stat[(size_t)b * Tn + tbase + tid];
    if (tid == 0) __builtin_amdgcn_s_wait_tensorcnt(0);   // tile `tile` landed
    __syncthreads();
    if (tid == 0 && tile + 1 < ntiles) {                  // prefetch next tile
      const int nb = (tile + 1) & 1;
      tdm_load_tile((unsigned int)(size_t)(etile + (size_t)nb * ETILE_FLOATS),
                    (unsigned long long)(size_t)(lbase + tbase + TILE_T));
    }
    const float* eb = etile + (size_t)(tile & 1) * ETILE_FLOATS;

    for (int tl = 0; tl < TILE_T; ++tl) {
      const int t = tbase + tl;
      const float st   = stt[tl];
      const float raw0 = eb[tl];                          // class 0 (blank)
      const float rawt = eb[toks[tid] * LROW + tl];       // gathered token class
      const float val  = fmaxf(pA[jj] + raw0, pA[jj - 1] + rawt) - st;
      float* orow = tre + (size_t)(t + 1) * (Ln + 1);
      pB[jj]  = val;
      orow[jj] = val;
      if (tid == 0) {
        c0 += raw0 - st;                                  // col0[t+1] cumsum
        const float c0v = ((t + 1) <= (Tn - Ln)) ? c0 : INFINITY;
        pB[0]  = c0v;
        orow[0] = c0v;
      }
      __syncthreads();
      float* tmp = pA; pA = pB; pB = tmp;
    }
  }
}

// ---------------------------------------------------------------------------
// Kernel 3: backtrack.  Parallel first-max argmax over trellis[:, L], then a
// single-lane T-step masked walk (matches the reference scan exactly).
// int/bool outputs written as floats.
// ---------------------------------------------------------------------------
__global__ __launch_bounds__(128) void fa_backtrack(
    const float* __restrict__ logits, const int* __restrict__ tokens,
    const float* __restrict__ stat, const float* __restrict__ trellis,
    float* __restrict__ pj, float* __restrict__ pt,
    float* __restrict__ pp, float* __restrict__ pv) {
  const int b   = blockIdx.x;
  const int tid = threadIdx.x;
  const float* tre = trellis + (size_t)b * (Tn + 1) * (Ln + 1);
  __shared__ float sv[128];
  __shared__ int   si[128];

  float bv = -INFINITY;
  int   bi = 0x7FFFFFFF;
  for (int i = tid; i <= Tn; i += 128) {
    const float v = tre[(size_t)i * (Ln + 1) + Ln];
    if (v > bv || (v == bv && i < bi)) { bv = v; bi = i; }
  }
  sv[tid] = bv; si[tid] = bi;
  __syncthreads();
  for (int s = 64; s > 0; s >>= 1) {
    if (tid < s) {
      const float v = sv[tid + s];
      const int   i = si[tid + s];
      if (v > sv[tid] || (v == sv[tid] && i < si[tid])) { sv[tid] = v; si[tid] = i; }
    }
    __syncthreads();
  }
  if (tid != 0) return;

  const int t_start = si[0];
  const float* lbase = logits + (size_t)b * Cn * Tn;
  const float* stb   = stat + (size_t)b * Tn;
  const int*   tokb  = tokens + (size_t)b * Ln;
  float* pjb = pj + (size_t)b * Tn;
  float* ptb = pt + (size_t)b * Tn;
  float* ppb = pp + (size_t)b * Tn;
  float* pvb = pv + (size_t)b * Tn;

  int  j    = Ln;
  bool done = false;
  for (int k = 0; k < Tn; ++k) {
    const int  t      = t_start - k;
    const bool active = (t >= 1) && !done;
    const int  tm1    = (t - 1 > 0) ? t - 1 : 0;
    const int  jm1    = (j - 1 > 0) ? j - 1 : 0;
    const float st  = stb[tm1];
    const float e0  = lbase[tm1] - st;                         // emission[tm1, 0]
    const int   tok = tokb[jm1];
    const float et  = lbase[(size_t)tok * Tn + tm1] - st;      // emission[tm1, tok]
    const float stayed  = tre[(size_t)tm1 * (Ln + 1) + j]   + e0;
    const float changed = tre[(size_t)tm1 * (Ln + 1) + jm1] + et;
    const bool  take = changed > stayed;
    pjb[k] = (float)jm1;
    ptb[k] = (float)tm1;
    ppb[k] = expf(take ? et : e0);
    pvb[k] = active ? 1.0f : 0.0f;
    if (active && take) j = j - 1;
    done = done || (j == 0) || (t <= 1);
  }
}

// ---------------------------------------------------------------------------
extern "C" void kernel_launch(void* const* d_in, const int* in_sizes, int n_in,
                              void* d_out, int out_size, void* d_ws, size_t ws_size,
                              hipStream_t stream) {
  const float* logits = (const float*)d_in[0];
  const int*   tokens = (const int*)d_in[1];
  float* stat = (float*)d_ws;                       // B*T floats (512 KB)
  float* out  = (float*)d_out;

  const size_t Ntr = (size_t)Bn * (Tn + 1) * (Ln + 1);
  float* trellis = out;
  float* pj = out + Ntr;
  float* pt = pj + (size_t)Bn * Tn;
  float* pp = pt + (size_t)Bn * Tn;
  float* pv = pp + (size_t)Bn * Tn;

  // 1) softmax stats: 16 rows/wave, 8 waves/block -> 128 rows/block
  fa_logsumexp_stats<<<dim3((Bn * Tn) / 128), dim3(256), 0, stream>>>(logits, stat);

  // 2) trellis DP: one workgroup per line, TDM-staged tiles in dynamic LDS
  const size_t smem = (size_t)(2 * 1026 + Ln + TILE_T) * 4 + (size_t)2 * ETILE_FLOATS * 4;
  fa_trellis<<<dim3(Bn), dim3(1024), smem, stream>>>(logits, tokens, stat, trellis);

  // 3) backtrack
  fa_backtrack<<<dim3(Bn), dim3(128), 0, stream>>>(logits, tokens, stat, trellis,
                                                   pj, pt, pp, pv);
}
